// LocalAttentionDraftLayer_25357486916438
// MI455X (gfx1250) — matrix operand order repaired and checked by reading, not compiled, not run on hardware
//
#include <hip/hip_runtime.h>
#include <math.h>

// ---------------------------------------------------------------------------
// Types for CDNA5 WMMA (wave32, 16x16x32 bf16 -> f32) and TDM descriptors
// ---------------------------------------------------------------------------
typedef __bf16 bf16_t;
typedef __attribute__((ext_vector_type(16))) __bf16   v16bf;
typedef __attribute__((ext_vector_type(8)))  float    v8f;
typedef __attribute__((ext_vector_type(4)))  unsigned u32x4;
typedef __attribute__((ext_vector_type(4)))  float    f32x4;
typedef __attribute__((ext_vector_type(4)))  unsigned tdm_g0_t;
typedef __attribute__((ext_vector_type(8)))  int      tdm_g1_t;
typedef __attribute__((ext_vector_type(4)))  int      tdm_g2_t;
typedef __attribute__((ext_vector_type(8)))  int      tdm_g4_t;

union bfvec { u32x4 u[2]; v16bf v; };

static constexpr int BB = 2;
static constexpr int SS = 4096;
static constexpr int HH = 1024;
static constexpr int WW = 32;
static constexpr int MM = BB * SS;   // 8192 rows

__device__ inline bf16_t f2bf(float f) {
  unsigned u = __builtin_bit_cast(unsigned, f);
  unsigned r = (u + 0x7FFFu + ((u >> 16) & 1u)) >> 16;   // round-to-nearest-even
  unsigned short h = (unsigned short)r;
  return __builtin_bit_cast(bf16_t, h);
}

__device__ inline v8f vzero8() { v8f z = {0.f,0.f,0.f,0.f,0.f,0.f,0.f,0.f}; return z; }

__device__ inline v8f wmma_bf16(v16bf a, v16bf b, v8f c) {
  return __builtin_amdgcn_wmma_f32_16x16x32_bf16(false, a, false, b, (short)0, c,
                                                 false, false);
}

// ---------------------------------------------------------------------------
// TDM: async 2D tile load (tile 32 cols x `tile_rows` rows of bf16) into LDS,
// with LDS padding 64B-data + 16B-pad -> 80B row stride (matches [..][40]).
// D# layout per CDNA5 ISA 8.3/8.4. Tracked by TENSORcnt.
// ---------------------------------------------------------------------------
__device__ inline void tdm_load_tile(unsigned lds_addr, const bf16_t* gptr,
                                     int tile_rows, int K) {
  unsigned long long ga = (unsigned long long)(const void*)gptr;
  tdm_g0_t g0;
  g0[0] = 1u;                                            // count=1 (valid user D#)
  g0[1] = lds_addr;                                      // lds_addr [63:32]
  g0[2] = (unsigned)(ga & 0xFFFFFFFFu);                  // global_addr lo
  g0[3] = (unsigned)((ga >> 32) & 0x01FFFFFFu) | (2u << 30);  // addr hi | type=2
  tdm_g1_t g1;
  g1[0] = (int)((1u << 16) |        // data_size = 2B
                (1u << 20) |        // pad_enable
                (3u << 22) |        // pad_interval: 16 DWORDs (64B)
                (3u << 25));        // pad_amount:   4 DWORDs (16B)
  g1[1] = (int)(((unsigned)K & 0xFFFFu) << 16);          // tensor_dim0[15:0]
  g1[2] = (int)(((unsigned)K >> 16) | (8192u << 16));    // td0 hi | tensor_dim1 lo
  g1[3] = (int)(32u << 16);                              // td1 hi=0 | tile_dim0=32
  g1[4] = (int)((unsigned)tile_rows & 0xFFFFu);          // tile_dim1 | tile_dim2=0
  g1[5] = (int)K;                                        // tensor_dim0_stride lo
  g1[6] = 0;                                             // stride hi | dim1_stride lo
  g1[7] = 0;
  tdm_g2_t z4 = {0, 0, 0, 0};
#if defined(__clang_major__) && (__clang_major__ >= 23)
  tdm_g4_t z8 = {0, 0, 0, 0, 0, 0, 0, 0};
  __builtin_amdgcn_tensor_load_to_lds(g0, g1, z4, z4, z8, 0);
#else
  __builtin_amdgcn_tensor_load_to_lds(g0, g1, z4, z4, 0);
#endif
}

// ---------------------------------------------------------------------------
// f32 -> bf16 conversion
// ---------------------------------------------------------------------------
__global__ __launch_bounds__(256) void cvt_f32_bf16(const float* __restrict__ in,
                                                    bf16_t* __restrict__ out, int n) {
  int i = blockIdx.x * blockDim.x + threadIdx.x;
  if (i < n) out[i] = f2bf(in[i]);
}

// ---------------------------------------------------------------------------
// Tiled bf16 WMMA GEMM:  C[M,N] = A[M,K] * Bw[N,K]^T   (+ epilogue)
// Block = 256 threads (8 waves), 128x128 tile, BK = 32, TDM double-buffered.
// Wave tile 32x64 (2x4 WMMA frags).
// Epilogues: 0 = scale -> bf16 (QKV)      1 = +resid -> f32 (Wo/draft)
//            2 = +bias, gelu -> bf16 (W1)  3 = +bias +resid -> f32 (W2/out)
// ---------------------------------------------------------------------------
template <int EP>
__global__ __launch_bounds__(256) void gemm_bf16_kernel(
    const bf16_t* __restrict__ A, const bf16_t* __restrict__ Bw,
    int M, int N, int K,
    const float* __restrict__ resid, const float* __restrict__ bias, float scale,
    float* __restrict__ outF, bf16_t* __restrict__ outB) {
  __shared__ bf16_t As[2][128][40];   // 80B row stride (TDM pad) -> aligned b128
  __shared__ bf16_t Bs[2][128][40];
  (void)M;

  const int bx = blockIdx.x, by = blockIdx.y;
  const int tid = threadIdx.x;
  const int lane = tid & 31, wv = tid >> 5;
  const int wm = (wv & 3) * 32;        // wave M offset in tile
  const int wn = (wv >> 2) * 64;       // wave N offset in tile
  const int l15 = lane & 15, lh = lane >> 4;

  const bf16_t* Abase = A + (size_t)(by * 128) * K;
  const bf16_t* Bbase = Bw + (size_t)(bx * 128) * K;

  v8f acc[2][4] = {vzero8(), vzero8(), vzero8(), vzero8(),
                   vzero8(), vzero8(), vzero8(), vzero8()};

  const int nk = K >> 5;
  if (wv == 0) {   // prologue: stage K-slice 0 into buffer 0
    tdm_load_tile((unsigned)(size_t)(void*)&As[0][0][0], Abase, 128, K);
    tdm_load_tile((unsigned)(size_t)(void*)&Bs[0][0][0], Bbase, 128, K);
  }

  int buf = 0;
  for (int t = 0; t < nk; ++t) {
    if (wv == 0) __builtin_amdgcn_s_wait_tensorcnt(0);
    __syncthreads();                       // buffer `buf` is ready for all waves
    if (wv == 0 && (t + 1) < nk) {         // async-stage next K-slice
      tdm_load_tile((unsigned)(size_t)(void*)&As[buf ^ 1][0][0],
                    Abase + (t + 1) * 32, 128, K);
      tdm_load_tile((unsigned)(size_t)(void*)&Bs[buf ^ 1][0][0],
                    Bbase + (t + 1) * 32, 128, K);
    }

    bfvec af[2], bfr[4];
#pragma unroll
    for (int i = 0; i < 2; i++) {          // A operand: K-split per lane-half
      int r = wm + i * 16 + l15;
      af[i].u[0] = *(const u32x4*)&As[buf][r][lh * 8];
      af[i].u[1] = *(const u32x4*)&As[buf][r][16 + lh * 8];
    }
#pragma unroll
    for (int j = 0; j < 4; j++) {          // B operand: K-split per lane-half
      int c = wn + j * 16 + l15;
      bfr[j].u[0] = *(const u32x4*)&Bs[buf][c][lh * 8];
      bfr[j].u[1] = *(const u32x4*)&Bs[buf][c][16 + lh * 8];
    }
#pragma unroll
    for (int i = 0; i < 2; i++)
#pragma unroll
      for (int j = 0; j < 4; j++)
        acc[i][j] = wmma_bf16(af[i].v, bfr[j].v, acc[i][j]);

    __syncthreads();                       // done reading `buf` before it is reused
    buf ^= 1;
  }

#pragma unroll
  for (int i = 0; i < 2; i++)
#pragma unroll
    for (int j = 0; j < 4; j++)
#pragma unroll
      for (int r = 0; r < 8; r++) {
        int gm = by * 128 + wm + i * 16 + lh * 8 + r;
        int gn = bx * 128 + wn + j * 16 + l15;
        size_t idx = (size_t)gm * N + gn;
        float v = acc[i][j][r];
        if constexpr (EP == 0) {
          outB[idx] = f2bf(v * scale);
        } else if constexpr (EP == 1) {
          outF[idx] = resid[idx] + v;
        } else if constexpr (EP == 2) {
          float tt = v + bias[gn];
          outB[idx] = f2bf(0.5f * tt * (1.0f + erff(tt * 0.70710678118654752f)));
        } else {
          outF[idx] = resid[idx] + v + bias[gn];
        }
      }
}

// ---------------------------------------------------------------------------
// Banded local attention: one wave per 16-query tile; 48-key band.
// Q is pre-scaled by 1/sqrt(H). Output bf16 (input to the Wo GEMM).
// ---------------------------------------------------------------------------
__global__ __launch_bounds__(128) void attn_local_kernel(
    const bf16_t* __restrict__ Q, const bf16_t* __restrict__ Kmat,
    const bf16_t* __restrict__ V, bf16_t* __restrict__ O) {
  __shared__ float sP[4][16][64];   // per-wave probs, padded to 64 cols

  const int lane = threadIdx.x & 31, wv = threadIdx.x >> 5;
  const int gt = blockIdx.x * 4 + wv;       // tile id, 0..511
  const int b = gt >> 8;                    // SS/16 = 256 tiles per batch
  const int t = gt & 255;
  const int qbase = t * 16;
  const int kbase = qbase - 32;             // band covers [kbase, kbase+47]
  const int l15 = lane & 15, lh = lane >> 4;
  const size_t boff = (size_t)b * SS * HH;
  const bf16_t* Qb = Q + boff;
  const bf16_t* Kb = Kmat + boff;
  const bf16_t* Vb = V + boff;

  // ---- scores: S = Q K^T over H, three 16-key tiles ----
  v8f sacc[3] = {vzero8(), vzero8(), vzero8()};
  const int qr = qbase + l15;
  for (int ks = 0; ks < HH; ks += 32) {
    bfvec af;
    const bf16_t* qp = Qb + (size_t)qr * HH + ks;
    af.u[0] = *(const u32x4*)(qp + lh * 8);
    af.u[1] = *(const u32x4*)(qp + 16 + lh * 8);
#pragma unroll
    for (int ti = 0; ti < 3; ti++) {
      int kr = kbase + ti * 16 + l15;
      kr = kr < 0 ? 0 : (kr > SS - 1 ? SS - 1 : kr);   // OOB rows get masked later
      const bf16_t* kp = Kb + (size_t)kr * HH + ks + lh * 8;
      bfvec bv;
      bv.u[0] = *(const u32x4*)kp;
      bv.u[1] = *(const u32x4*)(kp + 16);
      sacc[ti] = wmma_bf16(af.v, bv.v, sacc[ti]);
    }
  }

  // ---- dump scores to LDS ----
#pragma unroll
  for (int ti = 0; ti < 3; ti++)
#pragma unroll
    for (int r = 0; r < 8; r++)
      sP[wv][lh * 8 + r][ti * 16 + l15] = sacc[ti][r];
  if (lane < 16) {
#pragma unroll
    for (int c = 48; c < 64; c++) sP[wv][lane][c] = 0.f;
  }
  __syncthreads();

  // ---- masked softmax, one row per lane (lanes 0..15) ----
  if (lane < 16) {
    const int row = lane, qi = qbase + row;
    float mx = -1e30f;
    for (int c = 0; c < 48; c++) {
      int kj = kbase + c;
      bool valid = (kj >= 0) && (kj <= qi) && (kj > qi - WW);
      float s = valid ? sP[wv][row][c] : -1e30f;
      sP[wv][row][c] = s;
      mx = fmaxf(mx, s);
    }
    float sum = 0.f;
    for (int c = 0; c < 48; c++) {
      float e = expf(sP[wv][row][c] - mx);
      sP[wv][row][c] = e;
      sum += e;
    }
    float inv = 1.0f / sum;
    for (int c = 0; c < 48; c++) sP[wv][row][c] *= inv;
  }
  __syncthreads();

  // ---- P fragments (16x64 padded -> two 16x32 A operands) ----
  bfvec pf[2];
#pragma unroll
  for (int c = 0; c < 2; c++) {
    const float* pr = &sP[wv][l15][c * 32];
    v16bf pv;
#pragma unroll
    for (int e = 0; e < 8; e++) pv[e] = f2bf(pr[lh * 16 + e]);
#pragma unroll
    for (int e = 0; e < 8; e++) pv[e + 8] = f2bf(pr[lh * 16 + 8 + e]);
    pf[c].v = pv;
  }

  // ---- O = P V over the 64-key (padded) band ----
  for (int ht = 0; ht < HH / 16; ht++) {
    v8f oacc = vzero8();
#pragma unroll
    for (int c = 0; c < 2; c++) {
      v16bf bv;
      int col = ht * 16 + l15;
#pragma unroll
      for (int e = 0; e < 16; e++) {
        int row = kbase + c * 32 + lh * 16 + e;
        row = row < 0 ? 0 : (row > SS - 1 ? SS - 1 : row);  // P==0 where clamped
        bv[e] = Vb[(size_t)row * HH + col];
      }
      oacc = wmma_bf16(pf[c].v, bv, oacc);
    }
#pragma unroll
    for (int r = 0; r < 8; r++) {
      int gm = qbase + lh * 8 + r;
      O[boff + (size_t)gm * HH + ht * 16 + l15] = f2bf(oacc[r]);
    }
  }
}

// ---------------------------------------------------------------------------
// LayerNorm over H=1024, one block per row, output bf16
// ---------------------------------------------------------------------------
__global__ __launch_bounds__(256) void ln_kernel(const float* __restrict__ x,
                                                 const float* __restrict__ w,
                                                 const float* __restrict__ bb,
                                                 bf16_t* __restrict__ out) {
  __shared__ float rs[256], rq[256];
  const int row = blockIdx.x, tid = threadIdx.x;
  const float* xr = x + (size_t)row * HH;
  f32x4 v = *(const f32x4*)(xr + tid * 4);
  float s = v[0] + v[1] + v[2] + v[3];
  float q = v[0]*v[0] + v[1]*v[1] + v[2]*v[2] + v[3]*v[3];
  rs[tid] = s; rq[tid] = q;
  __syncthreads();
  for (int off = 128; off > 0; off >>= 1) {
    if (tid < off) { rs[tid] += rs[tid + off]; rq[tid] += rq[tid + off]; }
    __syncthreads();
  }
  const float mu = rs[0] * (1.0f / HH);
  const float var = rq[0] * (1.0f / HH) - mu * mu;
  const float inv = rsqrtf(var + 1e-5f);
#pragma unroll
  for (int e = 0; e < 4; e++) {
    int col = tid * 4 + e;
    float y = (v[e] - mu) * inv * w[col] + bb[col];
    out[(size_t)row * HH + col] = f2bf(y);
  }
}

// ---------------------------------------------------------------------------
// Launch
// ---------------------------------------------------------------------------
extern "C" void kernel_launch(void* const* d_in, const int* in_sizes, int n_in,
                              void* d_out, int out_size, void* d_ws, size_t ws_size,
                              hipStream_t stream) {
  (void)in_sizes; (void)n_in; (void)out_size; (void)ws_size;
  const float* x    = (const float*)d_in[0];
  const float* Wq   = (const float*)d_in[1];
  const float* Wk   = (const float*)d_in[2];
  const float* Wv   = (const float*)d_in[3];
  const float* Wo   = (const float*)d_in[4];
  const float* ln_w = (const float*)d_in[5];
  const float* ln_b = (const float*)d_in[6];
  const float* W1   = (const float*)d_in[7];
  const float* b1   = (const float*)d_in[8];
  const float* W2   = (const float*)d_in[9];
  const float* b2   = (const float*)d_in[10];

  char* p = (char*)d_ws;
  auto alloc = [&](size_t bytes) -> char* {
    char* r = p; p += (bytes + 255) & ~(size_t)255; return r;
  };
  bf16_t* xbf   = (bf16_t*)alloc((size_t)MM * HH * 2);
  bf16_t* wqb   = (bf16_t*)alloc((size_t)HH * HH * 2);
  bf16_t* wkb   = (bf16_t*)alloc((size_t)HH * HH * 2);
  bf16_t* wvb   = (bf16_t*)alloc((size_t)HH * HH * 2);
  bf16_t* wob   = (bf16_t*)alloc((size_t)HH * HH * 2);
  bf16_t* w1b   = (bf16_t*)alloc((size_t)(HH / 2) * HH * 2);
  bf16_t* w2b   = (bf16_t*)alloc((size_t)HH * (HH / 2) * 2);
  bf16_t* qbf   = (bf16_t*)alloc((size_t)MM * HH * 2);
  bf16_t* kbf   = (bf16_t*)alloc((size_t)MM * HH * 2);
  bf16_t* vbf   = (bf16_t*)alloc((size_t)MM * HH * 2);
  bf16_t* attnb = (bf16_t*)alloc((size_t)MM * HH * 2);
  float*  draft = (float*) alloc((size_t)MM * HH * 4);
  bf16_t* hlnb  = (bf16_t*)alloc((size_t)MM * HH * 2);
  bf16_t* gbf   = (bf16_t*)alloc((size_t)MM * (HH / 2) * 2);

  const int thr = 256;
  auto cvt = [&](const float* src, bf16_t* dst, int n) {
    cvt_f32_bf16<<<(n + thr - 1) / thr, thr, 0, stream>>>(src, dst, n);
  };
  cvt(x,  xbf, MM * HH);
  cvt(Wq, wqb, HH * HH);
  cvt(Wk, wkb, HH * HH);
  cvt(Wv, wvb, HH * HH);
  cvt(Wo, wob, HH * HH);
  cvt(W1, w1b, (HH / 2) * HH);
  cvt(W2, w2b, HH * (HH / 2));

  dim3 blk(256);
  dim3 gHH(HH / 128, MM / 128);     // N=1024 GEMMs
  dim3 gH2(HH / 256, MM / 128);     // N=512  GEMM

  const float qscale = 1.0f / 32.0f;   // 1/sqrt(H)
  gemm_bf16_kernel<0><<<gHH, blk, 0, stream>>>(xbf, wqb, MM, HH, HH,
                                               nullptr, nullptr, qscale, nullptr, qbf);
  gemm_bf16_kernel<0><<<gHH, blk, 0, stream>>>(xbf, wkb, MM, HH, HH,
                                               nullptr, nullptr, 1.0f, nullptr, kbf);
  gemm_bf16_kernel<0><<<gHH, blk, 0, stream>>>(xbf, wvb, MM, HH, HH,
                                               nullptr, nullptr, 1.0f, nullptr, vbf);

  attn_local_kernel<<<(BB * SS / 16) / 4, 128, 0, stream>>>(qbf, kbf, vbf, attnb);

  // draft = x + attn @ Wo^T
  gemm_bf16_kernel<1><<<gHH, blk, 0, stream>>>(attnb, wob, MM, HH, HH,
                                               x, nullptr, 1.0f, draft, nullptr);

  ln_kernel<<<MM, 256, 0, stream>>>(draft, ln_w, ln_b, hlnb);

  // g = gelu(h @ W1^T + b1)
  gemm_bf16_kernel<2><<<gH2, blk, 0, stream>>>(hlnb, w1b, MM, HH / 2, HH,
                                               nullptr, b1, 1.0f, nullptr, gbf);
  // out = draft + g @ W2^T + b2
  gemm_bf16_kernel<3><<<gHH, blk, 0, stream>>>(gbf, w2b, MM, HH, HH / 2,
                                               draft, b2, 1.0f, (float*)d_out, nullptr);
}
